// Attention_21706764714343
// MI455X (gfx1250) — matrix operand order repaired and checked by reading, not compiled, not run on hardware
//
#include <hip/hip_runtime.h>

typedef __attribute__((ext_vector_type(16))) _Float16 v16h;
typedef __attribute__((ext_vector_type(8)))  _Float16 v8h;
typedef __attribute__((ext_vector_type(4)))  _Float16 v4h;
typedef __attribute__((ext_vector_type(8)))  float    v8f;

static constexpr int BATCH = 8, SEQ = 1024, DIM = 1024, NH = 16, HD = 64;

union Frag { v16h v; v8h h[2]; };

// CDNA5 async global->LDS copy (16B per lane, tracked by ASYNCcnt).
// GVS addressing: mem = SGPR64 + VGPR32 + IOFFSET; LDS addr from VGPR.
__device__ __forceinline__ void async_copy_b128(unsigned lds_off,
                                                const void* base,
                                                unsigned gofs) {
    asm volatile("global_load_async_to_lds_b128 %0, %1, %2"
                 :: "v"(lds_off), "v"(gofs), "s"(base) : "memory");
}
__device__ __forceinline__ void wait_asynccnt0() {
    asm volatile("s_wait_asynccnt 0x0" ::: "memory");
}
__device__ __forceinline__ unsigned lds_addr(const void* p) {
    return (unsigned)(size_t)p;   // low 32 bits of generic LDS address = LDS offset
}

// ---------------------------------------------------------------------------
// fp32 -> f16 elementwise convert (x)
// ---------------------------------------------------------------------------
__global__ __launch_bounds__(256) void cvt_kernel(const float* __restrict__ in,
                                                  _Float16* __restrict__ out) {
    const int i = blockIdx.x * 256 + threadIdx.x;
    const float4 f = ((const float4*)in)[i];
    v4h h = {(_Float16)f.x, (_Float16)f.y, (_Float16)f.z, (_Float16)f.w};
    ((v4h*)out)[i] = h;
}

// ---------------------------------------------------------------------------
// fp32 W[K][N] -> f16 Wt[N][K] (64x64 LDS-tiled transpose)
// ---------------------------------------------------------------------------
__global__ __launch_bounds__(256) void transw_kernel(const float* __restrict__ W,
                                                     _Float16* __restrict__ Wt,
                                                     int N) {
    __shared__ _Float16 T[64 * 72];
    const int tid = threadIdx.x;
    const int n0 = blockIdx.x * 64, k0 = blockIdx.y * 64;
    for (int i = 0; i < 4; ++i) {
        int idx = tid + i * 256;
        int kr = idx >> 4, nc = (idx & 15) * 4;
        const float4 f = *(const float4*)(W + (size_t)(k0 + kr) * N + n0 + nc);
        T[(nc + 0) * 72 + kr] = (_Float16)f.x;
        T[(nc + 1) * 72 + kr] = (_Float16)f.y;
        T[(nc + 2) * 72 + kr] = (_Float16)f.z;
        T[(nc + 3) * 72 + kr] = (_Float16)f.w;
    }
    __syncthreads();
    for (int i = 0; i < 2; ++i) {
        int idx = tid + i * 256;
        int nr = idx >> 3, kc = (idx & 7) * 8;
        *(v8h*)(Wt + (size_t)(n0 + nr) * DIM + k0 + kc) = *(const v8h*)(T + nr * 72 + kc);
    }
}

// ---------------------------------------------------------------------------
// Tiled GEMM (all-f16 inputs): C[M,N] = A[M][1024] @ Wt[N][1024]^T + bias
// Async double-buffered LDS staging; 8 waves; tile 128x128, BK=32.
// MODE 0: epilogue scatters f16 q/k (row major [b,h,s,64]) and v transposed.
// MODE 1: epilogue writes fp32 out.
// ---------------------------------------------------------------------------
template <int MODE>
__global__ __launch_bounds__(256) void gemm_kernel(
    const _Float16* __restrict__ A, const _Float16* __restrict__ Wt,
    const float* __restrict__ bias,
    _Float16* __restrict__ q_ws, _Float16* __restrict__ k_ws,
    _Float16* __restrict__ v_t, float* __restrict__ out, int N)
{
    constexpr int BM = 128, BN = 128, BK = 32, AST = 40, BST = 40;
    __shared__ _Float16 As[2][BM * AST];
    __shared__ _Float16 Bs[2][BN * BST];

    const int tid = threadIdx.x, lane = tid & 31, w = tid >> 5;
    const int wm = w & 3, wn = w >> 2;
    const int m0 = blockIdx.y * BM, n0 = blockIdx.x * BN;

    v8f acc[2][4];
    for (int mi = 0; mi < 2; ++mi)
        for (int ni = 0; ni < 4; ++ni)
            acc[mi][ni] = (v8f){0.f,0.f,0.f,0.f,0.f,0.f,0.f,0.f};

    const int arow = lane & 15, akb = (lane >> 4) * 8;
    const int bkb  = (lane >> 4) * 8;

    // Each thread's copy job: 2 x 16B chunks of the A tile + 2 of the B tile.
    auto issue = [&](int buf, int k0) {
        for (int i = 0; i < 2; ++i) {
            int idx = tid + i * 256;            // 0..511
            int row = idx >> 2, c = idx & 3;    // 4 chunks per 32-half row
            async_copy_b128(lds_addr(&As[buf][row * AST + c * 8]), A,
                            (unsigned)(((size_t)(m0 + row) * DIM + k0 + c * 8) * 2));
            async_copy_b128(lds_addr(&Bs[buf][row * BST + c * 8]), Wt,
                            (unsigned)(((size_t)(n0 + row) * DIM + k0 + c * 8) * 2));
        }
    };

    issue(0, 0);
    for (int k0 = 0, it = 0; k0 < DIM; k0 += BK, ++it) {
        const int cur = it & 1;
        wait_asynccnt0();     // this wave's copies into buf `cur` done
        __syncthreads();      // everyone's copies done; prev buffer fully read
        if (k0 + BK < DIM) issue(cur ^ 1, k0 + BK);

        v16h af[2], bf[4];
        for (int mi = 0; mi < 2; ++mi) {
            const _Float16* p = &As[cur][(wm * 32 + mi * 16 + arow) * AST + akb];
            Frag u; u.h[0] = *(const v8h*)p; u.h[1] = *(const v8h*)(p + 16);
            af[mi] = u.v;
        }
        for (int ni = 0; ni < 4; ++ni) {
            const _Float16* p = &Bs[cur][(wn * 64 + ni * 16 + arow) * BST + bkb];
            Frag u; u.h[0] = *(const v8h*)p; u.h[1] = *(const v8h*)(p + 16);
            bf[ni] = u.v;
        }
        for (int mi = 0; mi < 2; ++mi)
            for (int ni = 0; ni < 4; ++ni)
                acc[mi][ni] = __builtin_amdgcn_wmma_f32_16x16x32_f16(
                    false, af[mi], false, bf[ni], (short)0, acc[mi][ni], false, false);
    }

    // ---- epilogue ----
    const int crow = (lane >> 4) << 3, ccol = lane & 15;
    for (int mi = 0; mi < 2; ++mi) {
        for (int ni = 0; ni < 4; ++ni) {
            const int nb = n0 + wn * 64 + ni * 16 + ccol;
            const float bv = bias[nb];
            for (int r = 0; r < 8; ++r) {
                const int m = m0 + wm * 32 + mi * 16 + crow + r;
                const float val = acc[mi][ni][r] + bv;
                if (MODE == 1) {
                    out[(size_t)m * N + nb] = val;
                } else {
                    const int b = m >> 10, s = m & 1023;
                    const int sec = nb >> 10, nn = nb & 1023;
                    const int h = nn >> 6, d = nn & 63;
                    if (sec == 0)
                        q_ws[((size_t)(b * NH + h) * SEQ + s) * HD + d] = (_Float16)val;
                    else if (sec == 1)
                        k_ws[((size_t)(b * NH + h) * SEQ + s) * HD + d] = (_Float16)val;
                    else
                        v_t[((size_t)(b * NH + h) * HD + d) * SEQ + s] = (_Float16)val;
                }
            }
        }
    }
}

// ---------------------------------------------------------------------------
// Causal flash attention. One wave owns a 16-query strip; 8 waves per block.
// Key blocks of 64. All 8 K-fragments (resp. V-fragments) are hoisted into
// distinct registers before the WMMA chain so the scheduler can clause the
// b128 loads and use partial loadcnt waits instead of full drains.
// ---------------------------------------------------------------------------
__global__ __launch_bounds__(256) void attn_kernel(
    const _Float16* __restrict__ q_ws, const _Float16* __restrict__ k_ws,
    const _Float16* __restrict__ v_t, _Float16* __restrict__ a_ws)
{
    constexpr int PST = 72;
    __shared__ _Float16 pbuf[8][16 * PST];

    const int tid = threadIdx.x, lane = tid & 31, w = tid >> 5;
    const int strip = blockIdx.x * 8 + w;
    const int qs = strip & (SEQ / 16 - 1);
    const int bh = strip >> 6;
    const int b = bh >> 4, h = bh & 15;
    const int qbase = qs * 16;

    const _Float16* Q = q_ws + (size_t)bh * SEQ * HD;
    const _Float16* K = k_ws + (size_t)bh * SEQ * HD;
    const _Float16* V = v_t + (size_t)bh * HD * SEQ;

    const int arow = lane & 15, akb = (lane >> 4) * 8;

    v16h qf[2];
    for (int c = 0; c < 2; ++c) {
        const _Float16* p = Q + (size_t)(qbase + arow) * HD + c * 32 + akb;
        Frag u; u.h[0] = *(const v8h*)p; u.h[1] = *(const v8h*)(p + 16);
        qf[c] = u.v;
    }

    float mrow[8], lrow[8];
    v8f o[4];
    for (int r = 0; r < 8; ++r) { mrow[r] = -3.0e38f; lrow[r] = 0.f; }
    for (int t = 0; t < 4; ++t) o[t] = (v8f){0.f,0.f,0.f,0.f,0.f,0.f,0.f,0.f};

    const int qtop = qbase + 15;
    for (int j0 = 0; j0 <= qtop; j0 += 64) {
        // ---- hoisted K fragments: 16 b128 loads in flight ----
        Frag ku[4][2];
        for (int t = 0; t < 4; ++t) {
            const int key = j0 + t * 16 + (lane & 15);
            const _Float16* p = K + (size_t)key * HD + ((lane >> 4) * 8);
            ku[t][0].h[0] = *(const v8h*)p;        ku[t][0].h[1] = *(const v8h*)(p + 16);
            ku[t][1].h[0] = *(const v8h*)(p + 32); ku[t][1].h[1] = *(const v8h*)(p + 48);
        }
        // ---- scores: four 16x16 C tiles (64 keys) ----
        v8f st[4];
        for (int t = 0; t < 4; ++t) {
            v8f z = (v8f){0.f,0.f,0.f,0.f,0.f,0.f,0.f,0.f};
            z = __builtin_amdgcn_wmma_f32_16x16x32_f16(false, qf[0], false, ku[t][0].v, (short)0, z, false, false);
            z = __builtin_amdgcn_wmma_f32_16x16x32_f16(false, qf[1], false, ku[t][1].v, (short)0, z, false, false);
            st[t] = z;
        }
        // ---- scale + causal mask ----
        for (int r = 0; r < 8; ++r) {
            const int qrow = qbase + ((lane >> 4) << 3) + r;
            for (int t = 0; t < 4; ++t) {
                float s = st[t][r] * 0.125f;
                if (j0 + t * 16 + (lane & 15) > qrow) s = -10000.f;
                st[t][r] = s;
            }
        }
        // ---- online softmax (per-row stats live per 16-lane group) ----
        for (int r = 0; r < 8; ++r) {
            float mv = fmaxf(fmaxf(st[0][r], st[1][r]), fmaxf(st[2][r], st[3][r]));
            for (int off = 1; off < 16; off <<= 1) mv = fmaxf(mv, __shfl_xor(mv, off, 32));
            const float mn = fmaxf(mrow[r], mv);
            const float sc = __expf(mrow[r] - mn);
            float su = 0.f;
            for (int t = 0; t < 4; ++t) {
                const float p = __expf(st[t][r] - mn);
                st[t][r] = p; su += p;
            }
            for (int off = 1; off < 16; off <<= 1) su += __shfl_xor(su, off, 32);
            lrow[r] = lrow[r] * sc + su;
            mrow[r] = mn;
            for (int t = 0; t < 4; ++t) o[t][r] *= sc;
        }
        // ---- bounce P through LDS: C layout -> A-fragment layout ----
        _Float16* pb = pbuf[w];
        const int prow = (lane >> 4) << 3, pcol = lane & 15;
        for (int t = 0; t < 4; ++t)
            for (int r = 0; r < 8; ++r)
                pb[(prow + r) * PST + t * 16 + pcol] = (_Float16)st[t][r];
        __builtin_amdgcn_wave_barrier();   // same-wave DS ops are in-order
        asm volatile("" ::: "memory");
        v16h pf[2];
        for (int c = 0; c < 2; ++c) {
            const _Float16* p = pb + (size_t)(lane & 15) * PST + c * 32 + akb;
            Frag u; u.h[0] = *(const v8h*)p; u.h[1] = *(const v8h*)(p + 16);
            pf[c] = u.v;
        }
        // ---- hoisted V fragments, then O += P @ V ----
        Frag vu[4][2];
        for (int t = 0; t < 4; ++t) {
            const int d = t * 16 + (lane & 15);
            const _Float16* pv = V + (size_t)d * SEQ + j0 + ((lane >> 4) * 8);
            vu[t][0].h[0] = *(const v8h*)pv;        vu[t][0].h[1] = *(const v8h*)(pv + 16);
            vu[t][1].h[0] = *(const v8h*)(pv + 32); vu[t][1].h[1] = *(const v8h*)(pv + 48);
        }
        for (int t = 0; t < 4; ++t) {
            o[t] = __builtin_amdgcn_wmma_f32_16x16x32_f16(false, pf[0], false, vu[t][0].v, (short)0, o[t], false, false);
            o[t] = __builtin_amdgcn_wmma_f32_16x16x32_f16(false, pf[1], false, vu[t][1].v, (short)0, o[t], false, false);
        }
    }

    // ---- normalize and store merged-head output (f16) ----
    for (int r = 0; r < 8; ++r) {
        const float inv = 1.0f / lrow[r];
        const int srow = qbase + ((lane >> 4) << 3) + r;
        for (int t = 0; t < 4; ++t) {
            const int col = h * HD + t * 16 + (lane & 15);
            a_ws[((size_t)(b * SEQ) + srow) * DIM + col] = (_Float16)(o[t][r] * inv);
        }
    }
}

extern "C" void kernel_launch(void* const* d_in, const int* in_sizes, int n_in,
                              void* d_out, int out_size, void* d_ws, size_t ws_size,
                              hipStream_t stream) {
    const float* x        = (const float*)d_in[0];
    const float* c_attn_w = (const float*)d_in[1];
    const float* c_attn_b = (const float*)d_in[2];
    const float* c_proj_w = (const float*)d_in[3];
    const float* c_proj_b = (const float*)d_in[4];

    const size_t n_tok  = (size_t)BATCH * SEQ * DIM;       // 8M
    const size_t n_qk   = (size_t)BATCH * NH * SEQ * HD;   // 8M
    _Float16* xh      = (_Float16*)d_ws;
    _Float16* wqkv_t  = xh + n_tok;                        // [3072][1024]
    _Float16* wproj_t = wqkv_t + (size_t)3 * DIM * DIM;    // [1024][1024]
    _Float16* q_ws    = wproj_t + (size_t)DIM * DIM;
    _Float16* k_ws    = q_ws + n_qk;
    _Float16* v_t     = k_ws + n_qk;
    _Float16* a_ws    = v_t + n_qk;

    // One-time f16 conversion / weight transposition (L2-resident afterwards)
    cvt_kernel<<<dim3(n_tok / 4 / 256), 256, 0, stream>>>(x, xh);
    transw_kernel<<<dim3(3 * DIM / 64, DIM / 64), 256, 0, stream>>>(c_attn_w, wqkv_t, 3 * DIM);
    transw_kernel<<<dim3(DIM / 64, DIM / 64), 256, 0, stream>>>(c_proj_w, wproj_t, DIM);

    // QKV: M=8192, N=3072
    gemm_kernel<0><<<dim3(3 * DIM / 128, BATCH * SEQ / 128), 256, 0, stream>>>(
        xh, wqkv_t, c_attn_b, q_ws, k_ws, v_t, nullptr, 3 * DIM);

    // Attention
    attn_kernel<<<dim3(BATCH * NH * (SEQ / 16) / 8), 256, 0, stream>>>(
        q_ws, k_ws, v_t, a_ws);

    // Projection: M=8192, N=1024
    gemm_kernel<1><<<dim3(DIM / 128, BATCH * SEQ / 128), 256, 0, stream>>>(
        a_ws, wproj_t, c_proj_b, nullptr, nullptr, nullptr, (float*)d_out, DIM);
}
